// GGMLLayer_33578054320687
// MI455X (gfx1250) — matrix operand order repaired and checked by reading, not compiled, not run on hardware
//
#include <hip/hip_runtime.h>

// ---------------------------------------------------------------------------
// y[m,o] = sum_k x[m,k] * ((q[o,k]-128)*scale[o,k/32]) + 4*(x@A^T@B^T)[m,o] + bias[o]
//   M = 4*256 = 1024, N = 8192, K = 8192
// HBM-bound on q_int (256 MB). Strategy:
//   k1: t = 2*(x @ lora_A^T)  -> [1024,16] f16 in d_ws
//   k2: tiled f16 WMMA GEMM; dequant q->f16 into LDS; rank-16 LoRA applied as
//       one extra zero-padded K=32 WMMA; bias in epilogue.
// ---------------------------------------------------------------------------

typedef __attribute__((ext_vector_type(16))) _Float16 v16h;
typedef __attribute__((ext_vector_type(8)))  _Float16 h8;
typedef __attribute__((ext_vector_type(8)))  float    v8f;

#define IN_F   8192
#define OUT_F  8192
#define RANK   16
#define M_TOT  1024
#define KBLK   32
#define LDSPAD 40   // halfs per row (32 data + 8 pad) -> 80B stride, conflict-free b128

union frag16 { v16h v; h8 h[2]; };

// ---------------- kernel 1: t[m,r] = 2 * sum_k x[m,k] * lora_A[r,k] ----------
__global__ __launch_bounds__(256) void lora_t_kernel(
    const float* __restrict__ x, const float* __restrict__ loraA,
    _Float16* __restrict__ t)
{
  int gid = blockIdx.x * blockDim.x + threadIdx.x;   // 1024*16 threads
  int m = gid >> 4;
  int r = gid & 15;
  const float4* xv = (const float4*)(x + (size_t)m * IN_F);
  const float4* av = (const float4*)(loraA + (size_t)r * IN_F);
  float acc = 0.f;
  #pragma unroll 4
  for (int i = 0; i < IN_F / 4; ++i) {
    float4 a = xv[i], b = av[i];
    acc = fmaf(a.x, b.x, acc); acc = fmaf(a.y, b.y, acc);
    acc = fmaf(a.z, b.z, acc); acc = fmaf(a.w, b.w, acc);
  }
  t[(size_t)m * RANK + r] = (_Float16)(2.0f * acc);
}

// ---------------- kernel 2: main dequant + WMMA GEMM -------------------------
__global__ __launch_bounds__(256) void ggml_lora_gemm_kernel(
    const float* __restrict__ x, const int* __restrict__ q,
    const float* __restrict__ scales, const float* __restrict__ loraB,
    const float* __restrict__ bias, const _Float16* __restrict__ t,
    float* __restrict__ out)
{
  __shared__ _Float16 xs [64  * LDSPAD];   // x tile, f16, row-major [m][k]
  __shared__ _Float16 wsm[128 * LDSPAD];   // dequant w tile, f16, [o][k]

  const int tid  = threadIdx.x;
  const int lane = tid & 31;
  const int wave = tid >> 5;           // 8 waves
  const int wm   = wave >> 2;          // 0..1 : wave row  (32 M each)
  const int wn   = wave & 3;           // 0..3 : wave col  (32 N each)
  const int m0   = blockIdx.y * 64;
  const int n0   = blockIdx.x * 128;
  const bool lo16 = (lane < 16);
  const int  lm   = lane & 15;

  // global-load work split (256 threads)
  const int xrow = tid >> 2, xq = tid & 3;   // x: 64 rows x (4 x 8 floats)
  const int wrow = tid >> 1, wh = tid & 1;   // q: 128 rows x (2 x 16 ints)

  const float* xg_base = x + (size_t)(m0 + xrow) * IN_F + xq * 8;
  const int*   qg_base = q + (size_t)(n0 + wrow) * IN_F + wh * 16;
  const float* sc_base = scales + (size_t)(n0 + wrow) * (IN_F / KBLK);

  v8f acc[2][2] = {};

  for (int kb = 0; kb < IN_F / KBLK; ++kb) {
    // ---- global loads first (latency overlap with barrier) ----
    const float4* xg = (const float4*)(xg_base + kb * KBLK);
    float4 xa = xg[0], xb = xg[1];
    const int4* qg = (const int4*)(qg_base + kb * KBLK);
    int4 q0 = qg[0], q1 = qg[1], q2 = qg[2], q3 = qg[3];
    float s = sc_base[kb];                 // one scale per (o, 32-block)

    __syncthreads();                       // previous tile fully consumed

    // ---- x -> f16 LDS ----
    h8 hx;
    hx[0]=(_Float16)xa.x; hx[1]=(_Float16)xa.y; hx[2]=(_Float16)xa.z; hx[3]=(_Float16)xa.w;
    hx[4]=(_Float16)xb.x; hx[5]=(_Float16)xb.y; hx[6]=(_Float16)xb.z; hx[7]=(_Float16)xb.w;
    *(h8*)&xs[xrow * LDSPAD + xq * 8] = hx;

    // ---- dequant q -> f16 LDS ----
    #define CVT(v_) ((_Float16)(((float)(v_) - 128.0f) * s))
    h8 hw0, hw1;
    hw0[0]=CVT(q0.x); hw0[1]=CVT(q0.y); hw0[2]=CVT(q0.z); hw0[3]=CVT(q0.w);
    hw0[4]=CVT(q1.x); hw0[5]=CVT(q1.y); hw0[6]=CVT(q1.z); hw0[7]=CVT(q1.w);
    hw1[0]=CVT(q2.x); hw1[1]=CVT(q2.y); hw1[2]=CVT(q2.z); hw1[3]=CVT(q2.w);
    hw1[4]=CVT(q3.x); hw1[5]=CVT(q3.y); hw1[6]=CVT(q3.z); hw1[7]=CVT(q3.w);
    #undef CVT
    *(h8*)&wsm[wrow * LDSPAD + wh * 16]     = hw0;
    *(h8*)&wsm[wrow * LDSPAD + wh * 16 + 8] = hw1;

    __syncthreads();

    // ---- fragment loads (contiguous k = 16*half + s, same for A and B) ----
    frag16 a[2], b[2];
    #pragma unroll
    for (int tm = 0; tm < 2; ++tm) {       // A 16x32 f16: K 0..15 / 16..31 per half-wave
      int base = (wm * 32 + tm * 16 + lm) * LDSPAD + (lo16 ? 0 : 16);
      a[tm].h[0] = *(const h8*)&xs[base];
      a[tm].h[1] = *(const h8*)&xs[base + 8];
    }
    #pragma unroll
    for (int tn = 0; tn < 2; ++tn) {       // B 32x16 f16: K 0..15 / 16..31 per half-wave
      int base = (wn * 32 + tn * 16 + lm) * LDSPAD + (lo16 ? 0 : 16);
      b[tn].h[0] = *(const h8*)&wsm[base];
      b[tn].h[1] = *(const h8*)&wsm[base + 8];
    }
    #pragma unroll
    for (int tm = 0; tm < 2; ++tm)
      #pragma unroll
      for (int tn = 0; tn < 2; ++tn)
        acc[tm][tn] = __builtin_amdgcn_wmma_f32_16x16x32_f16(
            false, a[tm].v, false, b[tn].v, (short)0, acc[tm][tn], false, false);
  }

  // ---- LoRA rank-16 update as one zero-padded K=32 WMMA step ----
  frag16 al[2], bl[2];
  #pragma unroll
  for (int tm = 0; tm < 2; ++tm) {
    int m = m0 + wm * 32 + tm * 16 + lm;
    al[tm].h[0] = *(const h8*)&t[(size_t)m * RANK];     // K 0..7
    al[tm].h[1] = *(const h8*)&t[(size_t)m * RANK + 8]; // K 8..15 (hi half x zero B)
  }
  const float lsc = lo16 ? 2.0f : 0.0f;    // lanes 16..31 carry K=16..31 -> zero
  #pragma unroll
  for (int tn = 0; tn < 2; ++tn) {
    int o = n0 + wn * 32 + tn * 16 + lm;
    const float4* bp = (const float4*)(loraB + (size_t)o * RANK);
    float4 b0 = bp[0], b1 = bp[1], b2 = bp[2], b3 = bp[3];
    bl[tn].h[0][0]=(_Float16)(lsc*b0.x); bl[tn].h[0][1]=(_Float16)(lsc*b0.y);
    bl[tn].h[0][2]=(_Float16)(lsc*b0.z); bl[tn].h[0][3]=(_Float16)(lsc*b0.w);
    bl[tn].h[0][4]=(_Float16)(lsc*b1.x); bl[tn].h[0][5]=(_Float16)(lsc*b1.y);
    bl[tn].h[0][6]=(_Float16)(lsc*b1.z); bl[tn].h[0][7]=(_Float16)(lsc*b1.w);
    bl[tn].h[1][0]=(_Float16)(lsc*b2.x); bl[tn].h[1][1]=(_Float16)(lsc*b2.y);
    bl[tn].h[1][2]=(_Float16)(lsc*b2.z); bl[tn].h[1][3]=(_Float16)(lsc*b2.w);
    bl[tn].h[1][4]=(_Float16)(lsc*b3.x); bl[tn].h[1][5]=(_Float16)(lsc*b3.y);
    bl[tn].h[1][6]=(_Float16)(lsc*b3.z); bl[tn].h[1][7]=(_Float16)(lsc*b3.w);
  }
  #pragma unroll
  for (int tm = 0; tm < 2; ++tm)
    #pragma unroll
    for (int tn = 0; tn < 2; ++tn)
      acc[tm][tn] = __builtin_amdgcn_wmma_f32_16x16x32_f16(
          false, al[tm].v, false, bl[tn].v, (short)0, acc[tm][tn], false, false);

  // ---- bias + store (C/D layout: lane -> col o; VGPR r -> row M=r or 8+r) ----
  #pragma unroll
  for (int tn = 0; tn < 2; ++tn) {
    int o = n0 + wn * 32 + tn * 16 + lm;
    float bv = bias[o];
    #pragma unroll
    for (int tm = 0; tm < 2; ++tm) {
      int mbase = m0 + wm * 32 + tm * 16 + (lo16 ? 0 : 8);
      #pragma unroll
      for (int r = 0; r < 8; ++r)
        out[(size_t)(mbase + r) * OUT_F + o] = acc[tm][tn][r] + bv;
    }
  }
}

// ---------------------------------------------------------------------------
extern "C" void kernel_launch(void* const* d_in, const int* in_sizes, int n_in,
                              void* d_out, int out_size, void* d_ws, size_t ws_size,
                              hipStream_t stream) {
  const float* x      = (const float*)d_in[0];   // [4,256,8192] f32
  const int*   q      = (const int*)  d_in[1];   // [8192,256,32] i32
  const float* scales = (const float*)d_in[2];   // [8192,256] f32
  const float* loraA  = (const float*)d_in[3];   // [16,8192] f32
  const float* loraB  = (const float*)d_in[4];   // [8192,16] f32
  const float* bias   = (const float*)d_in[5];   // [8192] f32
  float*       out    = (float*)d_out;           // [4,256,8192] f32
  _Float16*    t      = (_Float16*)d_ws;         // [1024,16] f16 (32 KB scratch)

  lora_t_kernel<<<(M_TOT * RANK) / 256, 256, 0, stream>>>(x, loraA, t);

  dim3 grid(OUT_F / 128, M_TOT / 64);            // 64 x 16 workgroups
  ggml_lora_gemm_kernel<<<grid, 256, 0, stream>>>(x, q, scales, loraB, bias, t, out);
}
